// EFF2_29145648070904
// MI455X (gfx1250) — matrix operand order repaired and checked by reading, not compiled, not run on hardware
//
#include <hip/hip_runtime.h>
#include <hip/hip_bf16.h>

// ---------------------------------------------------------------------------
// Types for CDNA5 WMMA (wave32): D = A(16x32 bf16) * B(32x16 bf16) + C(16x16 f32)
// ---------------------------------------------------------------------------
typedef __attribute__((ext_vector_type(16))) __bf16 v16bf;
typedef __attribute__((ext_vector_type(8)))  float  v8f;

__device__ __forceinline__ unsigned short f2bf(float f) {
    unsigned int u = __float_as_uint(f);
    unsigned int r = u + 0x7FFFu + ((u >> 16) & 1u);   // round-to-nearest-even
    return (unsigned short)(r >> 16);
}

// ---------------------------------------------------------------------------
// Weight convert to fragment-ready layout:
//   W[O=N, I=K] fp32  ->  Wf[k/32][Npad][32] bf16  (zero-padded n >= N)
// Inner 32 elements are K-within-block in plain k order, so a wave's
// B fragment (N = col, K = 16*(e/8) + hi*8 + e%8) is two contiguous b128 loads from global.
// ---------------------------------------------------------------------------
__global__ void wtrans_kernel(const float* __restrict__ W,
                              unsigned short* __restrict__ Wf,
                              int O, int I, int Npad) {
    int t = blockIdx.x * blockDim.x + threadIdx.x;
    if (t >= Npad * I) return;
    int n = t % Npad, k = t / Npad;
    float val = (n < O) ? W[(size_t)n * I + k] : 0.f;
    Wf[(((size_t)(k >> 5)) * Npad + n) * 32 + (k & 31)] = f2bf(val);
}

// BN folding: s = g*rsqrt(v+eps), b' = b - m*s
__global__ void bn_coef_kernel(const float* __restrict__ g, const float* __restrict__ b,
                               const float* __restrict__ m, const float* __restrict__ v,
                               float* __restrict__ s, float* __restrict__ bb, int n) {
    int i = blockIdx.x * blockDim.x + threadIdx.x;
    if (i >= n) return;
    float sc = g[i] * rsqrtf(v[i] + 1e-5f);
    s[i] = sc;
    bb[i] = b[i] - m[i] * sc;
}

// ---------------------------------------------------------------------------
// Depthwise 3x3 + BN + ReLU.  Input NCHW fp32 (virtual concat of in0|in1),
// output activation matrix [B*HW, C] bf16 (A operand of the pointwise GEMM).
// ---------------------------------------------------------------------------
__global__ void dw3x3_bn_relu_kernel(const float* __restrict__ in0,
                                     const float* __restrict__ in1,
                                     int C0, int C,
                                     const float* __restrict__ wdw,   // [C,9]
                                     const float* __restrict__ sc,
                                     const float* __restrict__ bb,
                                     unsigned short* __restrict__ out, // [B,HW,C] bf16
                                     int B, int H, int W) {
    const int HW = H * W;
    long long tid = (long long)blockIdx.x * blockDim.x + threadIdx.x;
    long long total = (long long)B * C * HW;
    if (tid >= total) return;
    int p = (int)(tid % HW);
    int c = (int)((tid / HW) % C);
    int b = (int)(tid / ((long long)HW * C));
    int px = p % W, py = p / W;
    const float* src = (c < C0) ? (in0 + ((size_t)b * C0 + c) * HW)
                                : (in1 + ((size_t)b * (C - C0) + (c - C0)) * HW);
    const float* wk = wdw + (size_t)c * 9;
    float acc = 0.f;
#pragma unroll
    for (int dy = -1; dy <= 1; ++dy) {
        int yy = py + dy;
        if (yy < 0 || yy >= H) continue;
#pragma unroll
        for (int dx = -1; dx <= 1; ++dx) {
            int xx = px + dx;
            if (xx < 0 || xx >= W) continue;
            acc += wk[(dy + 1) * 3 + (dx + 1)] * src[yy * W + xx];
        }
    }
    float v = acc * sc[c] + bb[c];
    v = v > 0.f ? v : 0.f;
    out[((size_t)b * HW + p) * C + c] = f2bf(v);
}

// ---------------------------------------------------------------------------
// WMMA GEMM:  Out = epi( A[M,K]bf16 * Wf[K/32][Npad][32]bf16 )
//   block tile 128x64, 8 wave32s (4M x 2N), wave tile 32x32
//   -> 4 x v_wmma_f32_16x16x32_bf16 per wave per K-step
//   A double-buffered through LDS (single barrier per K-step);
//   B fragments loaded directly from global (L2-resident weights, no LDS).
//   epilogue: out = act * s[n] + b[n]  (+ReLU) ; modes:
//     0: bf16 [M,N]   1: fp32 [M,N]   2: fp32 NCHW (+ optional NCHW residual)
// ---------------------------------------------------------------------------
#define BM 128
#define BN 64
#define BKK 32

__global__ __launch_bounds__(256)
void gemm_bf16_wmma_kernel(const unsigned short* __restrict__ A,
                           const unsigned short* __restrict__ Wf,
                           int M, int N, int Npad, int K,
                           const float* __restrict__ scl,   // may be null (=> 1)
                           const float* __restrict__ bia,   // may be null (=> 0)
                           int relu, int mode, int HW,
                           const float* __restrict__ resid, // NCHW residual or null
                           float* __restrict__ outF,
                           unsigned short* __restrict__ outH) {
    __shared__ __align__(16) unsigned short lA[2][BM * BKK];   // 2 x 8 KB ping-pong

    const int tid  = threadIdx.x;
    const int lane = tid & 31;
    const int wave = tid >> 5;
    const int hi   = lane >> 4;   // lane half
    const int ln   = lane & 15;
    const int wm   = wave >> 1;   // 0..3 -> 32 rows each
    const int wn   = wave & 1;    // 0..1 -> 32 cols each
    const int m0   = blockIdx.y * BM;
    const int n0   = blockIdx.x * BN;

    union V8 { v8f v; float f[8]; };
    V8 c00, c01, c10, c11;
#pragma unroll
    for (int r = 0; r < 8; ++r) {
        c00.f[r] = 0.f; c01.f[r] = 0.f; c10.f[r] = 0.f; c11.f[r] = 0.f;
    }

    // A staging assignment (loop-invariant): 32 contiguous bytes per thread
    const int arow = tid >> 1;            // 0..127
    const int acg  = (tid & 1) * 16;      // 0,16
    const unsigned short* aptr = A + (size_t)(m0 + arow) * K + acg;

    // B fragment base: column nfrag, K-block 0, element hi*8
    const int nfrag = n0 + wn * 32 + ln;
    const unsigned short* bptr = Wf + ((size_t)nfrag) * 32 + hi * 8;
    const size_t bstep = (size_t)Npad * 32;       // advance one K-block

    // prologue: stage K-block 0
    {
        *(uint4*)&lA[0][arow * BKK + acg]     = *(const uint4*)(aptr);
        *(uint4*)&lA[0][arow * BKK + acg + 8] = *(const uint4*)(aptr + 8);
    }
    __syncthreads();

    const int nsteps = K >> 5;
    for (int i = 0; i < nsteps; ++i) {
        const int cur = i & 1;
        // stage next K-block into idle buffer (overlaps with WMMAs below)
        if (i + 1 < nsteps) {
            const unsigned short* ap = aptr + (size_t)(i + 1) * BKK;
            *(uint4*)&lA[1 - cur][arow * BKK + acg]     = *(const uint4*)(ap);
            *(uint4*)&lA[1 - cur][arow * BKK + acg + 8] = *(const uint4*)(ap + 8);
            if (i + 2 < nsteps)
                __builtin_prefetch(ap + BKK, 0, 1);
        }
        // B fragments direct from global (fragment-ready layout, L2 resident)
        union VA { v16bf v; uint4 q[2]; } a0, a1, b0, b1;
        {
            const unsigned short* pb0 = bptr + (size_t)i * bstep;
            const unsigned short* pb1 = pb0 + 16 * 32;   // +16 columns
            b0.q[0] = *(const uint4*)(pb0);
            b0.q[1] = *(const uint4*)(pb0 + 16);
            b1.q[0] = *(const uint4*)(pb1);
            b1.q[1] = *(const uint4*)(pb1 + 16);
        }
        // A fragments from LDS (per ISA wave32 layout)
        {
            const unsigned short* pa0 = &lA[cur][(wm * 32 + ln) * BKK];
            a0.q[0] = *(const uint4*)(pa0 + hi * 8);
            a0.q[1] = *(const uint4*)(pa0 + hi * 8 + 16);
            const unsigned short* pa1 = &lA[cur][(wm * 32 + 16 + ln) * BKK];
            a1.q[0] = *(const uint4*)(pa1 + hi * 8);
            a1.q[1] = *(const uint4*)(pa1 + hi * 8 + 16);
        }
        c00.v = __builtin_amdgcn_wmma_f32_16x16x32_bf16(false, a0.v, false, b0.v,
                                                        (short)0, c00.v, false, false);
        c01.v = __builtin_amdgcn_wmma_f32_16x16x32_bf16(false, a0.v, false, b1.v,
                                                        (short)0, c01.v, false, false);
        c10.v = __builtin_amdgcn_wmma_f32_16x16x32_bf16(false, a1.v, false, b0.v,
                                                        (short)0, c10.v, false, false);
        c11.v = __builtin_amdgcn_wmma_f32_16x16x32_bf16(false, a1.v, false, b1.v,
                                                        (short)0, c11.v, false, false);
        __syncthreads();   // protects read of lA[cur] and staging of lA[1-cur]
    }

    // epilogue (C/D layout: N = ln, M = r + hi*8)
    const int nbase = n0 + wn * 32 + ln;
#pragma unroll
    for (int mi = 0; mi < 2; ++mi) {
#pragma unroll
        for (int f = 0; f < 2; ++f) {
            const int n = nbase + f * 16;
            if (n < N) {
                const float s  = scl ? scl[n] : 1.f;
                const float b_ = bia ? bia[n] : 0.f;
#pragma unroll
                for (int r = 0; r < 8; ++r) {
                    const int m = m0 + wm * 32 + mi * 16 + hi * 8 + r;
                    float val;
                    if (mi == 0) val = (f == 0 ? c00.f[r] : c01.f[r]);
                    else         val = (f == 0 ? c10.f[r] : c11.f[r]);
                    val = val * s + b_;
                    if (relu) val = val > 0.f ? val : 0.f;
                    if (mode == 0) {
                        outH[(size_t)m * N + n] = f2bf(val);
                    } else if (mode == 1) {
                        outF[(size_t)m * N + n] = val;
                    } else {
                        const int bb = m / HW, p = m % HW;
                        const size_t addr = ((size_t)bb * N + n) * HW + p;
                        if (resid) val += resid[addr];
                        outF[addr] = val;
                    }
                }
            }
        }
    }
}

// ---------------------------------------------------------------------------
// Deformable sampling: one wave per (b, q, head); lane = head-dim channel.
// off [M,64] (NH*NP*2), aw [M,32] (NH*NP, pre-softmax), v [M,256] fp32.
// px = qx + off_x, py = qy + off_y (align_corners=False identity simplification)
// Writes sampled [M,256] bf16 (A operand of the output-projection GEMM).
// ---------------------------------------------------------------------------
__global__ void msdeform_sample_kernel(const float* __restrict__ off,
                                       const float* __restrict__ awl,
                                       const float* __restrict__ v,
                                       unsigned short* __restrict__ sampled,
                                       int B, int H, int W) {
    const int HW = H * W;
    const int lane = threadIdx.x & 31;
    long long gw = ((long long)blockIdx.x * blockDim.x + threadIdx.x) >> 5;
    long long nw = (long long)B * HW * 8;
    if (gw >= nw) return;
    const int head = (int)(gw % 8);
    const int q    = (int)((gw / 8) % HW);
    const int b    = (int)(gw / (8LL * HW));
    const int qx = q % W, qy = q / W;
    const size_t rowq = (size_t)b * HW + q;

    const float* offp = off + rowq * 64 + head * 8;
    const float* awp  = awl + rowq * 32 + head * 4;
    const float a0 = awp[0], a1 = awp[1], a2 = awp[2], a3 = awp[3];
    const float mx = fmaxf(fmaxf(a0, a1), fmaxf(a2, a3));
    const float e0 = __expf(a0 - mx), e1 = __expf(a1 - mx);
    const float e2 = __expf(a2 - mx), e3 = __expf(a3 - mx);
    const float inv = 1.f / (e0 + e1 + e2 + e3);
    const float wts[4] = { e0 * inv, e1 * inv, e2 * inv, e3 * inv };

    const float* vb = v + (size_t)b * HW * 256;
    const int ch = head * 32 + lane;
    float acc = 0.f;
#pragma unroll
    for (int p = 0; p < 4; ++p) {
        const float px = (float)qx + offp[p * 2 + 0];
        const float py = (float)qy + offp[p * 2 + 1];
        const float x0f = floorf(px), y0f = floorf(py);
        const float tx = px - x0f, ty = py - y0f;
        const int x0 = (int)x0f, y0 = (int)y0f;
        float g00 = 0.f, g10 = 0.f, g01 = 0.f, g11 = 0.f;
        const bool xv0 = (x0 >= 0) & (x0 < W),     xv1 = (x0 + 1 >= 0) & (x0 + 1 < W);
        const bool yv0 = (y0 >= 0) & (y0 < H),     yv1 = (y0 + 1 >= 0) & (y0 + 1 < H);
        if (xv0 & yv0) g00 = vb[(size_t)(y0 * W + x0) * 256 + ch];
        if (xv1 & yv0) g10 = vb[(size_t)(y0 * W + x0 + 1) * 256 + ch];
        if (xv0 & yv1) g01 = vb[(size_t)((y0 + 1) * W + x0) * 256 + ch];
        if (xv1 & yv1) g11 = vb[(size_t)((y0 + 1) * W + x0 + 1) * 256 + ch];
        const float s = g00 * (1.f - tx) * (1.f - ty) + g10 * tx * (1.f - ty)
                      + g01 * (1.f - tx) * ty          + g11 * tx * ty;
        acc += wts[p] * s;
    }
    sampled[rowq * 256 + ch] = f2bf(acc);
}

// ---------------------------------------------------------------------------
// Host orchestration
// ---------------------------------------------------------------------------
extern "C" void kernel_launch(void* const* d_in, const int* in_sizes, int n_in,
                              void* d_out, int out_size, void* d_ws, size_t ws_size,
                              hipStream_t stream) {
    (void)in_sizes; (void)n_in; (void)out_size; (void)ws_size;
    const int B = 8, C = 256, H = 64, W = 64, HW = H * W;
    const int M = B * HW;  // 32768

    const float* x = (const float*)d_in[0];
    const float* y = (const float*)d_in[1];
    // dsconv params: q=2.., rv=12.., sc=22..  (dw_w,dw_g,dw_b,dw_m,dw_v,pw_w,pw_g,pw_b,pw_m,pw_v)
    const float* q_dw_w = (const float*)d_in[2];
    const float* rv_dw_w = (const float*)d_in[12];
    const float* sc_dw_w = (const float*)d_in[22];
    const float* q_pw_w = (const float*)d_in[7];
    const float* rv_pw_w = (const float*)d_in[17];
    const float* sc_pw_w = (const float*)d_in[27];
    // attention params: a1=32.., a2=40..  (so_w,so_b,aw_w,aw_b,vp_w,vp_b,op_w,op_b)
    const float* a1_so_w = (const float*)d_in[32]; const float* a1_so_b = (const float*)d_in[33];
    const float* a1_aw_w = (const float*)d_in[34]; const float* a1_aw_b = (const float*)d_in[35];
    const float* a1_vp_w = (const float*)d_in[36]; const float* a1_vp_b = (const float*)d_in[37];
    const float* a1_op_w = (const float*)d_in[38]; const float* a1_op_b = (const float*)d_in[39];
    const float* a2_so_w = (const float*)d_in[40]; const float* a2_so_b = (const float*)d_in[41];
    const float* a2_aw_w = (const float*)d_in[42]; const float* a2_aw_b = (const float*)d_in[43];
    const float* a2_vp_w = (const float*)d_in[44]; const float* a2_vp_b = (const float*)d_in[45];
    const float* a2_op_w = (const float*)d_in[46]; const float* a2_op_b = (const float*)d_in[47];

    // workspace layout
    char* ws = (char*)d_ws;
    size_t off = 0;
    auto alloc = [&](size_t bytes) -> char* {
        char* p = ws + off;
        off += (bytes + 255) & ~(size_t)255;
        return p;
    };
    // fragment-ready weights: [K/32][Npad][32] bf16
    unsigned short* wt_q   = (unsigned short*)alloc((size_t)512 * 256 * 2);  // K=512, Npad=256
    unsigned short* wt_rv  = (unsigned short*)alloc((size_t)256 * 256 * 2);
    unsigned short* wt_sc  = (unsigned short*)alloc((size_t)512 * 256 * 2);
    unsigned short* wt_so1 = (unsigned short*)alloc((size_t)256 * 64 * 2);   // Npad=64
    unsigned short* wt_aw1 = (unsigned short*)alloc((size_t)256 * 64 * 2);   // Npad=64 (N=32)
    unsigned short* wt_vp1 = (unsigned short*)alloc((size_t)256 * 256 * 2);
    unsigned short* wt_op1 = (unsigned short*)alloc((size_t)256 * 256 * 2);
    unsigned short* wt_so2 = (unsigned short*)alloc((size_t)256 * 64 * 2);
    unsigned short* wt_aw2 = (unsigned short*)alloc((size_t)256 * 64 * 2);
    unsigned short* wt_vp2 = (unsigned short*)alloc((size_t)256 * 256 * 2);
    unsigned short* wt_op2 = (unsigned short*)alloc((size_t)256 * 256 * 2);
    float* sb_qdw  = (float*)alloc(512 * 2 * 4);
    float* sb_qpw  = (float*)alloc(256 * 2 * 4);
    float* sb_rvdw = (float*)alloc(256 * 2 * 4);
    float* sb_rvpw = (float*)alloc(256 * 2 * 4);
    float* sb_scdw = (float*)alloc(512 * 2 * 4);
    float* sb_scpw = (float*)alloc(256 * 2 * 4);
    unsigned short* Adw     = (unsigned short*)alloc((size_t)M * 512 * 2);
    unsigned short* qf      = (unsigned short*)alloc((size_t)M * 256 * 2);
    unsigned short* valbf   = (unsigned short*)alloc((size_t)M * 256 * 2);
    unsigned short* sampled = (unsigned short*)alloc((size_t)M * 256 * 2);
    float* vmat  = (float*)alloc((size_t)M * 256 * 4);
    float* offb  = (float*)alloc((size_t)M * 64 * 4);
    float* awbuf = (float*)alloc((size_t)M * 32 * 4);
    float* rgb   = (float*)alloc((size_t)B * C * HW * 4);
    float* inf   = (float*)alloc((size_t)B * C * HW * 4);

    // ---- weight prep -------------------------------------------------------
    auto wt = [&](const float* Wm, unsigned short* Wf, int O, int I, int Npad) {
        int tot = Npad * I;
        wtrans_kernel<<<(tot + 255) / 256, 256, 0, stream>>>(Wm, Wf, O, I, Npad);
    };
    wt(q_pw_w, wt_q, 256, 512, 256);
    wt(rv_pw_w, wt_rv, 256, 256, 256);
    wt(sc_pw_w, wt_sc, 256, 512, 256);
    wt(a1_so_w, wt_so1, 64, 256, 64);   wt(a1_aw_w, wt_aw1, 32, 256, 64);
    wt(a1_vp_w, wt_vp1, 256, 256, 256); wt(a1_op_w, wt_op1, 256, 256, 256);
    wt(a2_so_w, wt_so2, 64, 256, 64);   wt(a2_aw_w, wt_aw2, 32, 256, 64);
    wt(a2_vp_w, wt_vp2, 256, 256, 256); wt(a2_op_w, wt_op2, 256, 256, 256);

    auto bn = [&](int base, float* sb, int n) {
        bn_coef_kernel<<<(n + 255) / 256, 256, 0, stream>>>(
            (const float*)d_in[base], (const float*)d_in[base + 1],
            (const float*)d_in[base + 2], (const float*)d_in[base + 3], sb, sb + n, n);
    };
    bn(3, sb_qdw, 512);  bn(8, sb_qpw, 256);
    bn(13, sb_rvdw, 256); bn(18, sb_rvpw, 256);
    bn(23, sb_scdw, 512); bn(28, sb_scpw, 256);

    auto dw = [&](const float* i0, const float* i1, int C0, int Ct, const float* wdw,
                  float* sb, unsigned short* out) {
        long long tot = (long long)B * Ct * HW;
        dw3x3_bn_relu_kernel<<<(unsigned)((tot + 255) / 256), 256, 0, stream>>>(
            i0, i1, C0, Ct, wdw, sb, sb + Ct, out, B, H, W);
    };
    auto gemm = [&](const unsigned short* A, const unsigned short* Wf, int N, int Npad,
                    int K, const float* s, const float* b, int relu, int mode,
                    const float* resid, float* oF, unsigned short* oH) {
        dim3 grid((N + BN - 1) / BN, M / BM);
        gemm_bf16_wmma_kernel<<<grid, 256, 0, stream>>>(A, Wf, M, N, Npad, K, s, b,
                                                        relu, mode, HW, resid, oF, oH);
    };
    auto sample = [&]() {
        long long waves = (long long)B * HW * 8;
        msdeform_sample_kernel<<<(unsigned)((waves * 32 + 255) / 256), 256, 0, stream>>>(
            offb, awbuf, vmat, sampled, B, H, W);
    };

    // ---- Q = dsconv(concat(x,y)) ------------------------------------------
    dw(x, y, 256, 512, q_dw_w, sb_qdw, Adw);
    gemm(Adw, wt_q, 256, 256, 512, sb_qpw, sb_qpw + 256, 1, 0, nullptr, nullptr, qf);

    // ---- branch 1: rv = dsconv(x); rgb = msdeform(qf, rv) + y --------------
    dw(x, x, 256, 256, rv_dw_w, sb_rvdw, Adw);
    gemm(Adw, wt_rv, 256, 256, 256, sb_rvpw, sb_rvpw + 256, 1, 0, nullptr, nullptr, valbf);
    gemm(valbf, wt_vp1, 256, 256, 256, nullptr, a1_vp_b, 0, 1, nullptr, vmat, nullptr);
    gemm(qf, wt_so1, 64, 64, 256, nullptr, a1_so_b, 0, 1, nullptr, offb, nullptr);
    gemm(qf, wt_aw1, 32, 64, 256, nullptr, a1_aw_b, 0, 1, nullptr, awbuf, nullptr);
    sample();
    gemm(sampled, wt_op1, 256, 256, 256, nullptr, a1_op_b, 0, 2, y, rgb, nullptr);

    // ---- branch 2: iv = dsconv(y) [RGB_V weights]; inf = msdeform + x ------
    dw(y, y, 256, 256, rv_dw_w, sb_rvdw, Adw);
    gemm(Adw, wt_rv, 256, 256, 256, sb_rvpw, sb_rvpw + 256, 1, 0, nullptr, nullptr, valbf);
    gemm(valbf, wt_vp2, 256, 256, 256, nullptr, a2_vp_b, 0, 1, nullptr, vmat, nullptr);
    gemm(qf, wt_so2, 64, 64, 256, nullptr, a2_so_b, 0, 1, nullptr, offb, nullptr);
    gemm(qf, wt_aw2, 32, 64, 256, nullptr, a2_aw_b, 0, 1, nullptr, awbuf, nullptr);
    sample();
    gemm(sampled, wt_op2, 256, 256, 256, nullptr, a2_op_b, 0, 2, x, inf, nullptr);

    // ---- out = dsconv(concat(rgb, inf)) -> d_out (NCHW fp32) ---------------
    dw(rgb, inf, 256, 512, sc_dw_w, sb_scdw, Adw);
    gemm(Adw, wt_sc, 256, 256, 512, sb_scpw, sb_scpw + 256, 1, 2, nullptr,
         (float*)d_out, nullptr);
}
